// TransformerBlock_9079560864165
// MI455X (gfx1250) — matrix operand order repaired and checked by reading, compiled, not run on hardware
//
#include <hip/hip_runtime.h>

// ---------------------------------------------------------------------------
// MI455X (gfx1250) transformer block: bf16 WMMA GEMMs + flash attention.
// Wave32, v_wmma_f32_16x16x32_bf16 everywhere, fp32 accumulate.
// ---------------------------------------------------------------------------

typedef __attribute__((ext_vector_type(16))) __bf16 v16bf;
typedef __attribute__((ext_vector_type(8)))  float  v8f;

constexpr int Bn   = 2;
constexpr int Sn   = 2048;
constexpr int Dn   = 2048;
constexpr int Hn   = 16;
constexpr int HDn  = 128;
constexpr int Gn   = 8;
constexpr int HIDn = 6144;

union Frag {
    v16bf v;
    unsigned int u[8];
    unsigned short h[16];
};

#define WMMA_BF16(a, b, c) \
    __builtin_amdgcn_wmma_f32_16x16x32_bf16(false, (a), false, (b), (short)0, (c), false, false)

// ---------------------------------------------------------------------------
// fp32 [K][N] -> bf16 [N][K] (transpose + convert) so WMMA B-fragments read
// contiguous K pairs per lane.
// ---------------------------------------------------------------------------
__global__ __launch_bounds__(256) void convert_transpose_kernel(
    const float* __restrict__ W, __bf16* __restrict__ Wt, int K, int N)
{
    __shared__ float tile[32][33];
    const int k0 = blockIdx.x * 32, n0 = blockIdx.y * 32;
    #pragma unroll
    for (int j = 0; j < 4; ++j)
        tile[threadIdx.y + j * 8][threadIdx.x] =
            W[(size_t)(k0 + threadIdx.y + j * 8) * N + n0 + threadIdx.x];
    __syncthreads();
    #pragma unroll
    for (int j = 0; j < 4; ++j)
        Wt[(size_t)(n0 + threadIdx.y + j * 8) * K + k0 + threadIdx.x] =
            (__bf16)tile[threadIdx.x][threadIdx.y + j * 8];
}

// ---------------------------------------------------------------------------
// RMSNorm over D=2048, fp32 in -> bf16 out. One block per token.
// ---------------------------------------------------------------------------
__global__ __launch_bounds__(256) void rmsnorm_kernel(
    const float* __restrict__ x, const float* __restrict__ scale,
    __bf16* __restrict__ out)
{
    const size_t off = (size_t)blockIdx.x * Dn;
    const int tid = threadIdx.x;
    float vals[8];
    float ss = 0.f;
    #pragma unroll
    for (int j = 0; j < 8; ++j) {
        vals[j] = x[off + tid + j * 256];
        ss += vals[j] * vals[j];
    }
    __shared__ float red[256];
    red[tid] = ss;
    __syncthreads();
    for (int o = 128; o > 0; o >>= 1) {
        if (tid < o) red[tid] += red[tid + o];
        __syncthreads();
    }
    const float rstd = rsqrtf(red[0] / (float)Dn + 1e-6f);
    #pragma unroll
    for (int j = 0; j < 8; ++j) {
        const int c = tid + j * 256;
        out[off + c] = (__bf16)(vals[j] * rstd * scale[c]);
    }
}

// ---------------------------------------------------------------------------
// Per-head RMSNorm (over HD=128) + RoPE, fp32 in [B,S,heads*HD] -> bf16 out.
// One 128-thread block per (b,s,head) row.
// ---------------------------------------------------------------------------
__global__ __launch_bounds__(128) void rmsnorm_rope_kernel(
    const float* __restrict__ in, const float* __restrict__ scale,
    const float* __restrict__ cosv, const float* __restrict__ sinv,
    __bf16* __restrict__ out, int heads)
{
    const int idx = blockIdx.x;
    const int s = (idx / heads) % Sn;
    const size_t off = (size_t)idx * HDn;
    const int d = threadIdx.x;
    const float x = in[off + d];
    __shared__ float red[128];
    red[d] = x * x;
    __syncthreads();
    for (int o = 64; o > 0; o >>= 1) {
        if (d < o) red[d] += red[d + o];
        __syncthreads();
    }
    const float rstd = rsqrtf(red[0] / (float)HDn + 1e-6f);
    const float xn = x * rstd * scale[d];
    const int pd = (d < 64) ? d + 64 : d - 64;
    const float xo = in[off + pd] * rstd * scale[pd];
    const float rot = (d < 64) ? -xo : xo;
    out[off + d] = (__bf16)(xn * cosv[s * HDn + d] + rot * sinv[s * HDn + d]);
}

// ---------------------------------------------------------------------------
// bf16 WMMA GEMM: C[M,N] = A[M,K] @ Bt[N,K]^T, fp32 accumulate.
// 256 threads = 8 waves; block tile 128x128; wave tile 64x32 (4x2 frags).
// MODE 0: bf16 out.  MODE 1: fp32 out.  MODE 2: fp32 out + residual.
// ---------------------------------------------------------------------------
template <int MODE>
__global__ __launch_bounds__(256) void gemm_kernel(
    const __bf16* __restrict__ A, const __bf16* __restrict__ Bt,
    __bf16* __restrict__ Cb, float* __restrict__ Cf,
    const float* __restrict__ resid, int M, int N, int K)
{
    const int lane = threadIdx.x & 31;
    const int wave = threadIdx.x >> 5;
    const int m0 = blockIdx.y * 128 + (wave & 1) * 64;
    const int n0 = blockIdx.x * 128 + (wave >> 1) * 32;
    const int lrow = lane & 15;
    const int half = lane >> 4;

    v8f acc[4][2];
    #pragma unroll
    for (int i = 0; i < 4; ++i)
        #pragma unroll
        for (int j = 0; j < 2; ++j) acc[i][j] = (v8f)0.f;

    const __bf16* arow[4];
    #pragma unroll
    for (int mi = 0; mi < 4; ++mi)
        arow[mi] = A + (size_t)(m0 + mi * 16 + lrow) * K + 16 * half;
    const __bf16* brow0 = Bt + (size_t)(n0 + lrow) * K + 16 * half;
    const __bf16* brow1 = Bt + (size_t)(n0 + 16 + lrow) * K + 16 * half;

    for (int k0 = 0; k0 < K; k0 += 32) {
        __builtin_prefetch(brow0 + k0 + 512, 0, 1);  // global_prefetch_b8
        __builtin_prefetch(brow1 + k0 + 512, 0, 1);
        Frag b0, b1;
        #pragma unroll
        for (int p = 0; p < 8; ++p) {
            b0.u[p] = *(const unsigned int*)(brow0 + k0 + 2 * p);
            b1.u[p] = *(const unsigned int*)(brow1 + k0 + 2 * p);
        }
        #pragma unroll
        for (int mi = 0; mi < 4; ++mi) {
            Frag a;
            #pragma unroll
            for (int p = 0; p < 4; ++p) {
                a.u[p]     = *(const unsigned int*)(arow[mi] + k0 + 2 * p);
                a.u[p + 4] = *(const unsigned int*)(arow[mi] + k0 + 8 + 2 * p);
            }
            acc[mi][0] = WMMA_BF16(a.v, b0.v, acc[mi][0]);
            acc[mi][1] = WMMA_BF16(a.v, b1.v, acc[mi][1]);
        }
    }

    #pragma unroll
    for (int mi = 0; mi < 4; ++mi)
        #pragma unroll
        for (int ni = 0; ni < 2; ++ni)
            #pragma unroll
            for (int r = 0; r < 8; ++r) {
                const int m = m0 + mi * 16 + r + 8 * half;
                const int n = n0 + ni * 16 + lrow;
                const size_t o = (size_t)m * N + n;
                const float v = acc[mi][ni][r];
                if constexpr (MODE == 0) Cb[o] = (__bf16)v;
                else if constexpr (MODE == 1) Cf[o] = v;
                else Cf[o] = v + resid[o];
            }
}

// ---------------------------------------------------------------------------
// Fused gate/up GEMM: ff = silu(A@Wg) * (A@Wu), bf16 out.
// ---------------------------------------------------------------------------
__global__ __launch_bounds__(256) void gemm_gateup_kernel(
    const __bf16* __restrict__ A, const __bf16* __restrict__ Bg,
    const __bf16* __restrict__ Bu, __bf16* __restrict__ Cb,
    int M, int N, int K)
{
    const int lane = threadIdx.x & 31;
    const int wave = threadIdx.x >> 5;
    const int m0 = blockIdx.y * 128 + (wave & 1) * 64;
    const int n0 = blockIdx.x * 128 + (wave >> 1) * 32;
    const int lrow = lane & 15;
    const int half = lane >> 4;

    v8f accg[4][2], accu[4][2];
    #pragma unroll
    for (int i = 0; i < 4; ++i)
        #pragma unroll
        for (int j = 0; j < 2; ++j) { accg[i][j] = (v8f)0.f; accu[i][j] = (v8f)0.f; }

    const __bf16* arow[4];
    #pragma unroll
    for (int mi = 0; mi < 4; ++mi)
        arow[mi] = A + (size_t)(m0 + mi * 16 + lrow) * K + 16 * half;
    const __bf16* bg0 = Bg + (size_t)(n0 + lrow) * K + 16 * half;
    const __bf16* bg1 = Bg + (size_t)(n0 + 16 + lrow) * K + 16 * half;
    const __bf16* bu0 = Bu + (size_t)(n0 + lrow) * K + 16 * half;
    const __bf16* bu1 = Bu + (size_t)(n0 + 16 + lrow) * K + 16 * half;

    for (int k0 = 0; k0 < K; k0 += 32) {
        __builtin_prefetch(bg0 + k0 + 512, 0, 1);
        __builtin_prefetch(bu0 + k0 + 512, 0, 1);
        Frag g0, g1, u0, u1;
        #pragma unroll
        for (int p = 0; p < 8; ++p) {
            g0.u[p] = *(const unsigned int*)(bg0 + k0 + 2 * p);
            g1.u[p] = *(const unsigned int*)(bg1 + k0 + 2 * p);
            u0.u[p] = *(const unsigned int*)(bu0 + k0 + 2 * p);
            u1.u[p] = *(const unsigned int*)(bu1 + k0 + 2 * p);
        }
        #pragma unroll
        for (int mi = 0; mi < 4; ++mi) {
            Frag a;
            #pragma unroll
            for (int p = 0; p < 4; ++p) {
                a.u[p]     = *(const unsigned int*)(arow[mi] + k0 + 2 * p);
                a.u[p + 4] = *(const unsigned int*)(arow[mi] + k0 + 8 + 2 * p);
            }
            accg[mi][0] = WMMA_BF16(a.v, g0.v, accg[mi][0]);
            accg[mi][1] = WMMA_BF16(a.v, g1.v, accg[mi][1]);
            accu[mi][0] = WMMA_BF16(a.v, u0.v, accu[mi][0]);
            accu[mi][1] = WMMA_BF16(a.v, u1.v, accu[mi][1]);
        }
    }

    #pragma unroll
    for (int mi = 0; mi < 4; ++mi)
        #pragma unroll
        for (int ni = 0; ni < 2; ++ni)
            #pragma unroll
            for (int r = 0; r < 8; ++r) {
                const int m = m0 + mi * 16 + r + 8 * half;
                const int n = n0 + ni * 16 + lrow;
                const float g = accg[mi][ni][r];
                const float u = accu[mi][ni][r];
                const float sg = g / (1.f + __expf(-g));
                Cb[(size_t)m * N + n] = (__bf16)(sg * u);
            }
}

// ---------------------------------------------------------------------------
// Flash attention, causal, GQA (H=16 over G=8 KV heads).
// One wave per 16-query tile; online softmax; QK^T and PV via WMMA bf16.
// q: [B,S,H*HD] bf16; k,v: [B,S,G*HD] bf16; ctx out: [B,S,H*HD] bf16.
// ---------------------------------------------------------------------------
__global__ __launch_bounds__(32) void attn_kernel(
    const __bf16* __restrict__ qm, const __bf16* __restrict__ km,
    const __bf16* __restrict__ vm, __bf16* __restrict__ ctx)
{
    const int lane = threadIdx.x & 31;
    const int half = lane >> 4;
    const int lcol = lane & 15;
    const int qt = blockIdx.x, h = blockIdx.y, b = blockIdx.z;
    const int qbase = qt * 16;
    const int g = h >> 1;                 // H/G = 2
    const int qld = Hn * HDn;             // 2048
    const int kld = Gn * HDn;             // 1024

    // Q tile 16x128 as four A-fragments (K-steps of 32)
    v16bf qa[4];
    {
        const __bf16* qrow =
            qm + ((size_t)(b * Sn + qbase + lcol)) * qld + h * HDn + 16 * half;
        #pragma unroll
        for (int kk = 0; kk < 4; ++kk) {
            Frag f;
            #pragma unroll
            for (int p = 0; p < 4; ++p) {
                f.u[p]     = *(const unsigned int*)(qrow + kk * 32 + 2 * p);
                f.u[p + 4] = *(const unsigned int*)(qrow + kk * 32 + 8 + 2 * p);
            }
            qa[kk] = f.v;
        }
    }

    float mrow[8], lrow[8];
    v8f cacc[8];
    #pragma unroll
    for (int r = 0; r < 8; ++r) { mrow[r] = -1e30f; lrow[r] = 0.f; }
    #pragma unroll
    for (int ni = 0; ni < 8; ++ni) cacc[ni] = (v8f)0.f;

    __shared__ __bf16 pl[16][32];

    const int kend = qbase + 16;
    const int nsteps = (kend + 31) / 32;
    const float sscale = 0.08838834764831845f;  // 1/sqrt(128)

    for (int st = 0; st < nsteps; ++st) {
        const int kv = st * 32;
        #pragma unroll
        for (int sub = 0; sub < 2; ++sub) {
            const int kbase = kv + sub * 16;
            int keyn = kbase + lcol;
            if (keyn > Sn - 1) keyn = Sn - 1;
            // scores tile 16x16: S = Q @ K^T
            v8f sc = (v8f)0.f;
            const __bf16* krow =
                km + ((size_t)(b * Sn + keyn)) * kld + g * HDn + 16 * half;
            #pragma unroll
            for (int kk = 0; kk < 4; ++kk) {
                Frag f;
                #pragma unroll
                for (int p = 0; p < 8; ++p)
                    f.u[p] = *(const unsigned int*)(krow + kk * 32 + 2 * p);
                sc = WMMA_BF16(qa[kk], f.v, sc);
            }
            // causal mask + online softmax (per 8 rows held by this lane-half)
            #pragma unroll
            for (int r = 0; r < 8; ++r) {
                const int row = qbase + r + 8 * half;
                float s = sc[r] * sscale;
                if (kbase + lcol > row) s = -1e30f;
                float t = s;
                #pragma unroll
                for (int msk = 8; msk >= 1; msk >>= 1)
                    t = fmaxf(t, __shfl_xor(t, msk, 32));
                const float newm = fmaxf(mrow[r], t);
                const float corr = __expf(mrow[r] - newm);
                const float p = __expf(s - newm);
                float ps = p;
                #pragma unroll
                for (int msk = 8; msk >= 1; msk >>= 1)
                    ps += __shfl_xor(ps, msk, 32);
                lrow[r] = lrow[r] * corr + ps;
                mrow[r] = newm;
                #pragma unroll
                for (int ni = 0; ni < 8; ++ni) cacc[ni][r] *= corr;
                pl[r + 8 * half][sub * 16 + lcol] = (__bf16)p;
            }
        }
        __syncthreads();
        // P (C-layout) -> A-fragment via LDS
        Frag pf;
        {
            const __bf16* prow = &pl[lcol][8 * half];
            #pragma unroll
            for (int p = 0; p < 4; ++p) {
                pf.u[p]     = *(const unsigned int*)(prow + 2 * p);
                pf.u[p + 4] = *(const unsigned int*)(prow + 16 + 2 * p);
            }
        }
        // ctx += P @ V   (8 d-tiles of 16)
        #pragma unroll
        for (int ni = 0; ni < 8; ++ni) {
            Frag vf;
            #pragma unroll
            for (int e = 0; e < 16; ++e) {
                int key = kv + 8 * half + (e & 7) + ((e >> 3) << 4);
                if (key > Sn - 1) key = Sn - 1;
                vf.h[e] = *(const unsigned short*)(vm +
                    ((size_t)(b * Sn + key)) * kld + g * HDn + ni * 16 + lcol);
            }
            cacc[ni] = WMMA_BF16(pf.v, vf.v, cacc[ni]);
        }
        __syncthreads();
    }

    #pragma unroll
    for (int ni = 0; ni < 8; ++ni)
        #pragma unroll
        for (int r = 0; r < 8; ++r) {
            const int row = qbase + r + 8 * half;
            const int d = ni * 16 + lcol;
            ctx[((size_t)(b * Sn + row)) * qld + h * HDn + d] =
                (__bf16)(cacc[ni][r] / lrow[r]);
        }
}

// ---------------------------------------------------------------------------
// Host: full block pipeline on `stream`.
// ---------------------------------------------------------------------------
extern "C" void kernel_launch(void* const* d_in, const int* in_sizes, int n_in,
                              void* d_out, int out_size, void* d_ws, size_t ws_size,
                              hipStream_t stream)
{
    (void)in_sizes; (void)n_in; (void)out_size; (void)ws_size;

    const float* x     = (const float*)d_in[0];
    // d_in[1] = mask (causality computed analytically)
    const float* cosv  = (const float*)d_in[2];
    const float* sinv  = (const float*)d_in[3];
    const float* Wq    = (const float*)d_in[4];
    const float* Wk    = (const float*)d_in[5];
    const float* Wv    = (const float*)d_in[6];
    const float* Wo    = (const float*)d_in[7];
    const float* qsc   = (const float*)d_in[8];
    const float* ksc   = (const float*)d_in[9];
    const float* n1s   = (const float*)d_in[10];
    const float* n2s   = (const float*)d_in[11];
    const float* Wg    = (const float*)d_in[12];
    const float* Wu    = (const float*)d_in[13];
    const float* Wd    = (const float*)d_in[14];
    float* out = (float*)d_out;

    char* ws = (char*)d_ws;
    size_t off = 0;
    auto alloc = [&](size_t bytes) -> void* {
        void* p = ws + off;
        off += (bytes + 255) & ~(size_t)255;
        return p;
    };

    const int M = Bn * Sn;  // 4096 tokens

    __bf16* h1   = (__bf16*)alloc((size_t)M * Dn * 2);
    __bf16* Wq_t = (__bf16*)alloc((size_t)Dn * (Hn * HDn) * 2);
    __bf16* Wk_t = (__bf16*)alloc((size_t)Dn * (Gn * HDn) * 2);
    __bf16* Wv_t = (__bf16*)alloc((size_t)Dn * (Gn * HDn) * 2);
    __bf16* Wo_t = (__bf16*)alloc((size_t)(Hn * HDn) * Dn * 2);
    __bf16* Wg_t = (__bf16*)alloc((size_t)Dn * HIDn * 2);
    __bf16* Wu_t = (__bf16*)alloc((size_t)Dn * HIDn * 2);
    __bf16* Wd_t = (__bf16*)alloc((size_t)HIDn * Dn * 2);
    float*  qf   = (float*)alloc((size_t)M * (Hn * HDn) * 4);
    float*  kf   = (float*)alloc((size_t)M * (Gn * HDn) * 4);
    __bf16* vb   = (__bf16*)alloc((size_t)M * (Gn * HDn) * 2);
    __bf16* qb   = (__bf16*)alloc((size_t)M * (Hn * HDn) * 2);
    __bf16* kb   = (__bf16*)alloc((size_t)M * (Gn * HDn) * 2);
    __bf16* ctxb = (__bf16*)alloc((size_t)M * (Hn * HDn) * 2);
    float*  ax   = (float*)alloc((size_t)M * Dn * 4);
    __bf16* h2   = (__bf16*)alloc((size_t)M * Dn * 2);
    __bf16* ffb  = (__bf16*)alloc((size_t)M * HIDn * 2);

    const dim3 tb(32, 8);
    convert_transpose_kernel<<<dim3(Dn / 32, (Hn * HDn) / 32), tb, 0, stream>>>(Wq, Wq_t, Dn, Hn * HDn);
    convert_transpose_kernel<<<dim3(Dn / 32, (Gn * HDn) / 32), tb, 0, stream>>>(Wk, Wk_t, Dn, Gn * HDn);
    convert_transpose_kernel<<<dim3(Dn / 32, (Gn * HDn) / 32), tb, 0, stream>>>(Wv, Wv_t, Dn, Gn * HDn);
    convert_transpose_kernel<<<dim3((Hn * HDn) / 32, Dn / 32), tb, 0, stream>>>(Wo, Wo_t, Hn * HDn, Dn);
    convert_transpose_kernel<<<dim3(Dn / 32, HIDn / 32), tb, 0, stream>>>(Wg, Wg_t, Dn, HIDn);
    convert_transpose_kernel<<<dim3(Dn / 32, HIDn / 32), tb, 0, stream>>>(Wu, Wu_t, Dn, HIDn);
    convert_transpose_kernel<<<dim3(HIDn / 32, Dn / 32), tb, 0, stream>>>(Wd, Wd_t, HIDn, Dn);

    // h1 = rmsnorm(x) * norm1_scale  (bf16)
    rmsnorm_kernel<<<M, 256, 0, stream>>>(x, n1s, h1);

    // Q/K/V projections
    gemm_kernel<1><<<dim3((Hn * HDn) / 128, M / 128), 256, 0, stream>>>(
        h1, Wq_t, nullptr, qf, nullptr, M, Hn * HDn, Dn);
    gemm_kernel<1><<<dim3((Gn * HDn) / 128, M / 128), 256, 0, stream>>>(
        h1, Wk_t, nullptr, kf, nullptr, M, Gn * HDn, Dn);
    gemm_kernel<0><<<dim3((Gn * HDn) / 128, M / 128), 256, 0, stream>>>(
        h1, Wv_t, vb, nullptr, nullptr, M, Gn * HDn, Dn);

    // per-head rmsnorm + RoPE -> bf16
    rmsnorm_rope_kernel<<<M * Hn, 128, 0, stream>>>(qf, qsc, cosv, sinv, qb, Hn);
    rmsnorm_rope_kernel<<<M * Gn, 128, 0, stream>>>(kf, ksc, cosv, sinv, kb, Gn);

    // flash attention
    attn_kernel<<<dim3(Sn / 16, Hn, Bn), 32, 0, stream>>>(qb, kb, vb, ctxb);

    // attn output projection + residual
    gemm_kernel<2><<<dim3(Dn / 128, M / 128), 256, 0, stream>>>(
        ctxb, Wo_t, nullptr, ax, x, M, Dn, Hn * HDn);

    // FFN
    rmsnorm_kernel<<<M, 256, 0, stream>>>(ax, n2s, h2);
    gemm_gateup_kernel<<<dim3(HIDn / 128, M / 128), 256, 0, stream>>>(
        h2, Wg_t, Wu_t, ffb, M, HIDn, Dn);
    gemm_kernel<2><<<dim3(Dn / 128, M / 128), 256, 0, stream>>>(
        ffb, Wd_t, nullptr, out, ax, M, Dn, HIDn);
}
